// FaceClassifier_66400194396563
// MI455X (gfx1250) — matrix operand order repaired and checked by reading, not compiled, not run on hardware
//
#include <hip/hip_runtime.h>

// ---------------- problem constants (match reference) ----------------
constexpr int BMESH = 8;
constexpr int NPER  = 4096;
constexpr int KNN   = 16;
constexpr int NNODE = BMESH * NPER;      // 32768
constexpr int IN_DIM = 32;
constexpr int HID    = 128;
constexpr int NKE    = NNODE * KNN;      // 524288 directed edges per half
constexpr int ROWP   = HID + 4;          // LDS row pitch (floats), bank-conflict pad
constexpr float NEGINF = -3.0e38f;

typedef __attribute__((ext_vector_type(16))) __bf16 v16bf;
typedef __attribute__((ext_vector_type(8)))  float  v8f;

__device__ __forceinline__ v8f wmma_bf16(v16bf a, v16bf b, v8f c) {
  // D = A(16x32 bf16) * B(32x16 bf16) + C(16x16 f32)
  return __builtin_amdgcn_wmma_f32_16x16x32_bf16(
      false, a, false, b, (short)0, c, false, false);
}

// 16-bit A 16x32 — lane: M=lane&15; half=lane>>4;
// element e -> K = half*16 + e (same k order as B)
__device__ __forceinline__ int a_kmap(int lane, int e) {
  int half = (lane >> 4) & 1;
  return half * 16 + e;
}
// B 32x16: lane: N=lane&15; lanes 0-15 hold K=0..15, lanes 16-31 K=16..31
__device__ __forceinline__ int b_kmap(int lane, int e) {
  return ((lane >> 4) & 1) * 16 + e;
}

__device__ __forceinline__ void atomic_max_f32(float* p, float v) {
  __hip_atomic_fetch_max(p, v, __ATOMIC_RELAXED, __HIP_MEMORY_SCOPE_AGENT);
}

// CDNA5 async global->LDS copy, 16B per lane (GLOBAL_LOAD_ASYNC_TO_LDS_B128).
// ldsoff = byte offset in wave's LDS space; gaddr = 64-bit global address.
__device__ __forceinline__ void async_copy_b128(unsigned ldsoff,
                                                unsigned long long gaddr) {
  asm volatile("global_load_async_to_lds_b128 %0, %1, off"
               :: "v"(ldsoff), "v"(gaddr)
               : "memory");
}
__device__ __forceinline__ void wait_asynccnt0() {
  asm volatile("s_wait_asynccnt 0x0" ::: "memory");
}

// ---------------- 1) per-mesh KNN (K=16 smallest d^2) ----------------
__global__ __launch_bounds__(256) void knn_kernel(const float* __restrict__ pos,
                                                  int* __restrict__ nbr) {
  __shared__ float cpx[256], cpy[256], cpz[256];
  const int tid  = threadIdx.x;
  const int q    = blockIdx.x * 256 + tid;       // global node
  const int base = q & ~(NPER - 1);              // mesh start (NPER pow2)
  const int qloc = q & (NPER - 1);
  const float qx = pos[q * 3 + 0], qy = pos[q * 3 + 1], qz = pos[q * 3 + 2];

  float best[KNN];
  int   bidx[KNN];
#pragma unroll
  for (int j = 0; j < KNN; ++j) { best[j] = 3.0e38f; bidx[j] = qloc; }

  for (int t = 0; t < NPER / 256; ++t) {
    __syncthreads();
    const int c = t * 256 + tid;
    cpx[tid] = pos[(base + c) * 3 + 0];
    cpy[tid] = pos[(base + c) * 3 + 1];
    cpz[tid] = pos[(base + c) * 3 + 2];
    __syncthreads();
#pragma unroll 4
    for (int cc = 0; cc < 256; ++cc) {
      const int cl = t * 256 + cc;
      const float dx = qx - cpx[cc], dy = qy - cpy[cc], dz = qz - cpz[cc];
      const float d2 = dx * dx + dy * dy + dz * dz;
      if (cl != qloc && d2 < best[KNN - 1]) {
        float bd = d2; int bi = cl;
#pragma unroll
        for (int j = 0; j < KNN; ++j) {
          if (bd < best[j]) {
            float tb = best[j]; int ti = bidx[j];
            best[j] = bd; bidx[j] = bi; bd = tb; bi = ti;
          }
        }
      }
    }
  }
#pragma unroll
  for (int j = 0; j < KNN; ++j) nbr[q * KNN + j] = base + bidx[j];
}

// ---------------- 2) node GEMM: a = act(hin) @ W1[:D_IN] + b1 ----------------
// one wave per 16x16 output tile; grid = (N/16)*(HID/16) waves
template <int D_IN, int RELU>
__global__ __launch_bounds__(256) void node_gemm_kernel(
    const float* __restrict__ hin, const float* __restrict__ w1,
    const float* __restrict__ b1, float* __restrict__ aout) {
  const int lane = threadIdx.x & 31;
  const int wid  = (blockIdx.x * blockDim.x + threadIdx.x) >> 5;
  const int mt   = wid >> 3;       // node tile
  const int nt   = wid & 7;        // col tile
  const int mrow = mt * 16 + (lane & 15);
  const int col  = nt * 16 + (lane & 15);

  v8f acc = {};
#pragma unroll
  for (int ks = 0; ks < D_IN / 32; ++ks) {
    v16bf af, bf;
#pragma unroll
    for (int e = 0; e < 16; ++e) {
      int k = ks * 32 + a_kmap(lane, e);
      float v = hin[mrow * D_IN + k];
      if (RELU) v = fmaxf(v, 0.0f);
      af[e] = (__bf16)v;
    }
#pragma unroll
    for (int e = 0; e < 16; ++e) {
      int k = ks * 32 + b_kmap(lane, e);
      bf[e] = (__bf16)w1[k * HID + col];
    }
    acc = wmma_bf16(af, bf, acc);
  }
  const float bias = b1[col];
#pragma unroll
  for (int v = 0; v < 8; ++v) {
    int node = mt * 16 + v + ((lane >> 4) << 3);
    aout[node * HID + col] = acc[v] + bias;
  }
}

// ---------------- 3) fill h_next with -inf ----------------
__global__ __launch_bounds__(256) void fill_kernel(float4* __restrict__ p) {
  p[blockIdx.x * blockDim.x + threadIdx.x] =
      make_float4(NEGINF, NEGINF, NEGINF, NEGINF);
}

// ---------------- 4) edge kernel: m = relu(a[src]+pdiff*W1b) @ W2 + b2;
//                     h_next[dst] = max(h_next[dst], m)  ----------------
// one wave per 16 edges; 32 WMMAs (8 N-tiles x 4 K-steps) per wave.
// a[src] rows staged global->LDS via async B128 copies (coalesced, ASYNCcnt).
__global__ __launch_bounds__(256) void edge_kernel(
    const float* __restrict__ abuf, const float* __restrict__ pos,
    const int* __restrict__ nbr, const float* __restrict__ w1, int d_in,
    const float* __restrict__ w2, const float* __restrict__ b2,
    float* __restrict__ hout) {
  __shared__ __bf16 lw2[32 * 32 * 16];        // 32 B-fragments x 32 lanes x 16
  __shared__ float  lw1b[3 * HID];
  __shared__ float  lb2[HID];
  __shared__ float  lda[8 * 16 * ROWP];       // 8 waves x 16 rows (padded)
  const int tid = threadIdx.x;

  // stage W2 into LDS, pre-swizzled to WMMA B-fragment order
  for (int it = tid; it < 32 * 32 * 16; it += 256) {
    const int f = it >> 9, rem = it & 511;
    const int l = rem >> 4, e = rem & 15;
    const int ks = f >> 3, nt = f & 7;
    const int col = nt * 16 + (l & 15);
    const int k   = ks * 32 + ((l >> 4) << 4) + e;
    lw2[it] = (__bf16)w2[k * HID + col];
  }
  for (int it = tid; it < 3 * HID; it += 256)
    lw1b[it] = w1[(d_in + 6 + (it >> 7)) * HID + (it & 127)];
  if (tid < HID) lb2[tid] = b2[tid];
  __syncthreads();

  const int lane = tid & 31;
  const int half = lane >> 4;
  const int wv   = tid >> 5;
  const int wid  = (blockIdx.x << 3) + wv;           // wave-tile 0..65535
  const int ge0  = wid * 16;                         // < 2*NKE
  // wave-uniform direction flag, forced scalar (s_cbranch, no EXEC juggling)
  const int dir  = __builtin_amdgcn_readfirstlane((ge0 >= NKE) ? 1 : 0);
  const int e1   = (ge0 & (NKE - 1)) + (lane & 15);
  const int i    = e1 >> 4;                          // K = 16
  const int nb   = nbr[e1];
  const int src  = dir ? nb : i;
  const int dst  = dir ? i : nb;
  const float px = pos[src * 3 + 0] - pos[dst * 3 + 0];
  const float py = pos[src * 3 + 1] - pos[dst * 3 + 1];
  const float pz = pos[src * 3 + 2] - pos[dst * 3 + 2];

  // async-stage this wave's 16 a[src] rows: lane covers 16B of each 512B row.
  // Row's src is held in lane r (rows == lanes 0-15): v_readlane -> SGPR base.
  float* warow = &lda[wv * 16 * ROWP];
#pragma unroll
  for (int r = 0; r < 16; ++r) {
    const int s = __builtin_amdgcn_readlane(src, r);     // uniform
    const unsigned long long ga =
        (unsigned long long)(uintptr_t)(abuf + (size_t)s * HID) +
        (unsigned)(lane * 16);
    const unsigned lo =
        (unsigned)(uintptr_t)(warow + r * ROWP) + (unsigned)(lane * 16);
    async_copy_b128(lo, ga);
  }

  // dst for the C-layout scatter: lanes 0-15 output rows v, lanes 16-31 rows
  // v+8.  Two explicit readlanes (constant lane) + one cndmask per row —
  // NO variable indexing (avoids SGPR-array select trees).
  const int dall = __builtin_amdgcn_readlane(dst, 0);  // dir==1: shared dst
  int drow[8];
  if (!dir) {
#pragma unroll
    for (int v = 0; v < 8; ++v) {
      const int dlo = __builtin_amdgcn_readlane(dst, v);
      const int dhi = __builtin_amdgcn_readlane(dst, v + 8);
      drow[v] = half ? dhi : dlo;                      // one v_cndmask
    }
  }

  wait_asynccnt0();

  // build A fragments (16 edges x 128, bf16) for all 4 K-steps from LDS
  const float* myrow = warow + (lane & 15) * ROWP;
  v16bf afr[4];
#pragma unroll
  for (int ks = 0; ks < 4; ++ks) {
#pragma unroll
    for (int e = 0; e < 16; ++e) {
      const int k = ks * 32 + a_kmap(lane, e);
      float v = myrow[k] + px * lw1b[k] + py * lw1b[HID + k] +
                pz * lw1b[2 * HID + k];
      afr[ks][e] = (__bf16)fmaxf(v, 0.0f);
    }
  }

#pragma unroll
  for (int nt = 0; nt < 8; ++nt) {
    v8f acc = {};
#pragma unroll
    for (int ks = 0; ks < 4; ++ks) {
      const v16bf bf = *(const v16bf*)&lw2[(((ks * 8 + nt) * 32) + lane) * 16];
      acc = wmma_bf16(afr[ks], bf, acc);
    }
    const int col  = nt * 16 + (lane & 15);
    const float bb = lb2[col];
    if (dir) {
      // all 16 edges share dst: reduce rows in-wave, one atomic per column
      float mx = NEGINF;
#pragma unroll
      for (int v = 0; v < 8; ++v) mx = fmaxf(mx, acc[v] + bb);
      mx = fmaxf(mx, __shfl_xor(mx, 16, 32));  // lanes L and L+16 share col
      if (lane < 16) atomic_max_f32(&hout[dall * HID + col], mx);
    } else {
#pragma unroll
      for (int v = 0; v < 8; ++v) {
        atomic_max_f32(&hout[drow[v] * HID + col], acc[v] + bb);
      }
    }
  }
}

// ---------------- 5) logits = relu(h) @ w_final + b_final ----------------
__global__ __launch_bounds__(256) void logits_kernel(
    const float* __restrict__ h, const float* __restrict__ wf,
    const float* __restrict__ bf, float* __restrict__ logits) {
  __shared__ float lwf[HID];
  if (threadIdx.x < HID) lwf[threadIdx.x] = wf[threadIdx.x];
  __syncthreads();
  const int n = blockIdx.x * blockDim.x + threadIdx.x;
  const float4* hr = (const float4*)(h + (size_t)n * HID);
  float s = bf[0];
#pragma unroll
  for (int c = 0; c < HID / 4; ++c) {
    const float4 v = hr[c];
    s += fmaxf(v.x, 0.0f) * lwf[4 * c + 0] + fmaxf(v.y, 0.0f) * lwf[4 * c + 1] +
         fmaxf(v.z, 0.0f) * lwf[4 * c + 2] + fmaxf(v.w, 0.0f) * lwf[4 * c + 3];
  }
  logits[n] = s;
}

// ---------------- 6) per-mesh softmax ----------------
__global__ __launch_bounds__(256) void softmax_kernel(
    const float* __restrict__ logits, float* __restrict__ out) {
  __shared__ float red[256];
  const int tid = threadIdx.x;
  const float* L = logits + blockIdx.x * NPER;
  float* O = out + blockIdx.x * NPER;

  float mx = NEGINF;
  for (int i = tid; i < NPER; i += 256) mx = fmaxf(mx, L[i]);
  red[tid] = mx; __syncthreads();
  for (int s = 128; s > 0; s >>= 1) {
    if (tid < s) red[tid] = fmaxf(red[tid], red[tid + s]);
    __syncthreads();
  }
  const float M = red[0]; __syncthreads();

  float sum = 0.0f;
  for (int i = tid; i < NPER; i += 256) sum += __expf(L[i] - M);
  red[tid] = sum; __syncthreads();
  for (int s = 128; s > 0; s >>= 1) {
    if (tid < s) red[tid] += red[tid + s];
    __syncthreads();
  }
  const float inv = 1.0f / red[0];
  for (int i = tid; i < NPER; i += 256) O[i] = __expf(L[i] - M) * inv;
}

// ---------------- host launcher ----------------
extern "C" void kernel_launch(void* const* d_in, const int* in_sizes, int n_in,
                              void* d_out, int out_size, void* d_ws,
                              size_t ws_size, hipStream_t stream) {
  const float* x   = (const float*)d_in[0];
  const float* pos = (const float*)d_in[1];
  const float* w1[3] = {(const float*)d_in[3], (const float*)d_in[7],
                        (const float*)d_in[11]};
  const float* b1[3] = {(const float*)d_in[4], (const float*)d_in[8],
                        (const float*)d_in[12]};
  const float* w2[3] = {(const float*)d_in[5], (const float*)d_in[9],
                        (const float*)d_in[13]};
  const float* b2[3] = {(const float*)d_in[6], (const float*)d_in[10],
                        (const float*)d_in[14]};
  const float* wfin = (const float*)d_in[15];
  const float* bfin = (const float*)d_in[16];
  float* out = (float*)d_out;

  char* ws = (char*)d_ws;
  size_t off = 0;
  int*   nbr    = (int*)(ws + off);   off += (size_t)NNODE * KNN * 4;   // 2 MB
  float* abuf   = (float*)(ws + off); off += (size_t)NNODE * HID * 4;   // 16 MB
  float* h0     = (float*)(ws + off); off += (size_t)NNODE * HID * 4;   // 16 MB
  float* h1     = (float*)(ws + off); off += (size_t)NNODE * HID * 4;   // 16 MB
  float* logits = (float*)(ws + off);                                   // 128 KB

  knn_kernel<<<NNODE / 256, 256, 0, stream>>>(pos, nbr);

  const int gemm_blocks = (NNODE / 16) * (HID / 16) / 8;  // 8 waves/block
  float* houts[3] = {h0, h1, h0};
  const float* hin = x;
  int din = IN_DIM;
  for (int L = 0; L < 3; ++L) {
    if (L == 0)
      node_gemm_kernel<IN_DIM, 0><<<gemm_blocks, 256, 0, stream>>>(
          hin, w1[L], b1[L], abuf);
    else
      node_gemm_kernel<HID, 1><<<gemm_blocks, 256, 0, stream>>>(
          hin, w1[L], b1[L], abuf);
    fill_kernel<<<NNODE * HID / 4 / 256, 256, 0, stream>>>((float4*)houts[L]);
    // wave-tiles = 2*N*K/16 = 65536; 8 waves/block
    edge_kernel<<<2 * NKE / 16 / 8, 256, 0, stream>>>(
        abuf, pos, nbr, w1[L], din, w2[L], b2[L], houts[L]);
    hin = houts[L];
    din = HID;
  }

  logits_kernel<<<NNODE / 256, 256, 0, stream>>>(houts[2], wfin, bfin, logits);
  softmax_kernel<<<BMESH, 256, 0, stream>>>(logits, out);
}
